// DDiTBlock_UCUR_61323543052346
// MI455X (gfx1250) — matrix operand order repaired and checked by reading, not compiled, not run on hardware
//
#include <hip/hip_runtime.h>

typedef unsigned short u16;
typedef unsigned int   u32;
typedef __attribute__((ext_vector_type(16))) __bf16 v16bf;
typedef __attribute__((ext_vector_type(8)))  float  v8f;
typedef __attribute__((ext_vector_type(4)))  int    v4i;

#define DIMX   512
#define HEADS  8
#define SEQ    2048
#define BATCH  4
#define DH     64
#define NEGF   (-4294967295.0f)   // -2^32 + 1

union Frag16 { v16bf v; u16 u[16]; uint4 q[2]; };
union Frag8  { v8f  v; float f[8]; };

__device__ __forceinline__ u16 f2bf(float f) {
    u32 u = __float_as_uint(f);
    u32 r = (u + 0x7FFFu + ((u >> 16) & 1u)) >> 16;   // RNE
    return (u16)r;
}

// ---- gfx1250 async LDS DMA (GLOBAL_LOAD_ASYNC_TO_LDS_B128, ASYNCcnt) ------
#if defined(__has_builtin)
#if __has_builtin(__builtin_amdgcn_global_load_async_to_lds_b128)
#define HAVE_ASYNC_LDS_BUILTIN 1
#endif
#if __has_builtin(__builtin_amdgcn_s_wait_asynccnt)
#define HAVE_WAIT_ASYNC_BUILTIN 1
#endif
#endif

typedef __attribute__((address_space(1))) v4i gv4i;   // global int4
typedef __attribute__((address_space(3))) v4i lv4i;   // LDS int4

__device__ __forceinline__ void async_lds_load_b128(void* lds_dst, const void* gsrc) {
#ifdef HAVE_ASYNC_LDS_BUILTIN
    __builtin_amdgcn_global_load_async_to_lds_b128(
        (gv4i*)gsrc, (lv4i*)lds_dst, 0, 0);
#else
    // VDST = LDS byte address VGPR, VADDR = 64-bit global address (GV mode)
    asm volatile("global_load_async_to_lds_b128 %0, %1, off"
                 :: "v"((u32)(uintptr_t)lds_dst),
                    "v"((unsigned long long)(uintptr_t)gsrc)
                 : "memory");
#endif
}

__device__ __forceinline__ void wait_async0() {
#ifdef HAVE_WAIT_ASYNC_BUILTIN
    __builtin_amdgcn_s_wait_asynccnt(0);
#else
    asm volatile("s_wait_asynccnt 0x0" ::: "memory");
#endif
}

// block = 256 threads = 8 waves (wave32). Reduces two values across the block.
__device__ __forceinline__ void blockReduce2(float& a, float& b) {
    __shared__ float sa[8], sb[8];
    for (int m = 1; m < 32; m <<= 1) {
        a += __shfl_xor(a, m, 32);
        b += __shfl_xor(b, m, 32);
    }
    int w = threadIdx.x >> 5, l = threadIdx.x & 31;
    if (l == 0) { sa[w] = a; sb[w] = b; }
    __syncthreads();
    float ta = 0.f, tb = 0.f;
#pragma unroll
    for (int i = 0; i < 8; ++i) { ta += sa[i]; tb += sb[i]; }
    a = ta; b = tb;
    __syncthreads();
}

// ---------------------------------------------------------------------------
// f32 -> bf16 conversion (weights)
// ---------------------------------------------------------------------------
__global__ __launch_bounds__(256) void cvt_bf16_kernel(const float* __restrict__ in,
                                                       u16* __restrict__ out, int n) {
    int i = blockIdx.x * 256 + threadIdx.x;
    if (i < n) out[i] = f2bf(in[i]);
}

// ---------------------------------------------------------------------------
// RoPE + LN1 + masks. One block per (b,t) row of 512.
// ---------------------------------------------------------------------------
__global__ __launch_bounds__(256) void rope_ln1_kernel(
    const float* __restrict__ x, const float* __restrict__ ln1_w,
    float* __restrict__ qn, u16* __restrict__ qn_b, u16* __restrict__ emb_b,
    float* __restrict__ key_mask, float* __restrict__ query_mask) {
    int row = blockIdx.x;              // b*SEQ + t
    int t   = row & (SEQ - 1);
    int j   = threadIdx.x;             // 0..255
    const float* xr = x + (size_t)row * DIMX;
    float x0 = xr[j], x1 = xr[j + 256];
    float theta = __powf(10000.f, -(float)j * (1.f / 256.f));
    float s, c;
    __sincosf((float)t * theta, &s, &c);
    float e0 = x0 * c - x1 * s;        // first half: x*cos - x_hi*sin
    float e1 = x1 * c + x0 * s;        // second half: x*cos + x_lo*sin
    float sum = e0 + e1, sq = e0 * e0 + e1 * e1;
    blockReduce2(sum, sq);
    float ksum = sum;
    float mean = sum * (1.f / DIMX);
    float var  = sq * (1.f / DIMX) - mean * mean;
    float rstd = rsqrtf(var + 1e-5f);
    float q0 = (e0 - mean) * rstd * ln1_w[j];
    float q1 = (e1 - mean) * rstd * ln1_w[j + 256];
    float qsum = q0 + q1, dummy = 0.f;
    blockReduce2(qsum, dummy);
    size_t o = (size_t)row * DIMX + j;
    qn[o] = q0;  qn[o + 256] = q1;
    qn_b[o] = f2bf(q0);  qn_b[o + 256] = f2bf(q1);
    emb_b[o] = f2bf(e0); emb_b[o + 256] = f2bf(e1);
    if (j == 0) {
        key_mask[row]   = (ksum != 0.f) ? 1.f : 0.f;
        query_mask[row] = (qsum != 0.f) ? 1.f : 0.f;
    }
}

// ---------------------------------------------------------------------------
// GEMM: C[M,512] = A[M,512] * W[512,512]^T + bias (NT, bf16 in, f32 acc)
// block = 8 waves; each wave owns a 16x64 tile; block tile = 128x64.
// W k-chunks streamed with double-buffered GLOBAL_LOAD_ASYNC_TO_LDS_B128.
// ---------------------------------------------------------------------------
#define GM_OUT_QK   0   // bf16 out, layout [B,H,T,64]
#define GM_OUT_VT   1   // bf16 out, layout [B,H,64,T]   (V transposed)
#define GM_OUT_F32  2   // f32  out, [M,512]
#define GM_OUT_BF16 3   // bf16 out, [M,512]

__global__ __launch_bounds__(256) void gemm_nt_wmma(
    const u16* __restrict__ A, const u16* __restrict__ W,
    const float* __restrict__ bias, void* __restrict__ out,
    int mode, int relu) {
    // two buffers of 64 n-rows x 32 k (pad to 40 u16 -> 80B rows, 16B aligned)
    __shared__ u16 sW[2][64][40];
    const int w    = threadIdx.x >> 5;
    const int lane = threadIdx.x & 31;
    const int half = lane >> 4;
    const int ln   = lane & 15;
    const int m0   = blockIdx.x * 128 + w * 16;
    const int n0   = blockIdx.y * 64;

    const int rrow = threadIdx.x >> 2;          // 0..63 (one 16B chunk per lane)
    const int c8   = (threadIdx.x & 3) * 8;     // 0,8,16,24
    const u16* Wsrc = W + (size_t)(n0 + rrow) * DIMX + c8;

    const u16* Arow = A + (size_t)(m0 + ln) * DIMX;

    Frag8 acc[4];
#pragma unroll
    for (int i = 0; i < 4; ++i)
#pragma unroll
        for (int r = 0; r < 8; ++r) acc[i].f[r] = 0.f;

    // prime buffer 0
    async_lds_load_b128(&sW[0][rrow][c8], Wsrc);
    wait_async0();
    __syncthreads();

    for (int kk = 0; kk < DIMX; kk += 32) {
        const int buf = (kk >> 5) & 1;
        const bool more = (kk + 32) < DIMX;
        if (more)   // prefetch next W tile into the other buffer (ASYNCcnt)
            async_lds_load_b128(&sW[buf ^ 1][rrow][c8], Wsrc + kk + 32);

        if (more) __builtin_prefetch(Arow + kk + 32, 0, 1);
        Frag16 a;   // A fragment 16x32: k = (e>>3)*16 + half*8 + (e&7)
        a.q[0] = *(const uint4*)(Arow + kk + half * 8);
        a.q[1] = *(const uint4*)(Arow + kk + 16 + half * 8);
#pragma unroll
        for (int nt = 0; nt < 4; ++nt) {
            Frag16 bf;  // B fragment 32x16: k = (e>>3)*16 + half*8 + (e&7)
            const u16* bp = &sW[buf][nt * 16 + ln][half * 8];
            bf.q[0] = *(const uint4*)(bp);
            bf.q[1] = *(const uint4*)(bp + 16);
            acc[nt].v = __builtin_amdgcn_wmma_f32_16x16x32_bf16(
                false, a.v, false, bf.v, (short)0, acc[nt].v, false, false);
        }
        if (more) {
            wait_async0();       // our async fill of buf^1 has landed in LDS
            __syncthreads();     // all waves done reading buf & done filling
        }
    }
#pragma unroll
    for (int nt = 0; nt < 4; ++nt) {
        int n = n0 + nt * 16 + ln;
        float bv = bias[n];
#pragma unroll
        for (int r = 0; r < 8; ++r) {
            int m = m0 + r + half * 8;
            float v = acc[nt].f[r] + bv;
            if (relu) v = fmaxf(v, 0.f);
            if (mode == GM_OUT_QK) {
                int b = m / SEQ, t = m % SEQ, h = blockIdx.y, c = nt * 16 + ln;
                ((u16*)out)[((size_t)(b * HEADS + h) * SEQ + t) * DH + c] = f2bf(v);
            } else if (mode == GM_OUT_VT) {
                int b = m / SEQ, t = m % SEQ, h = blockIdx.y, c = nt * 16 + ln;
                ((u16*)out)[((size_t)(b * HEADS + h) * DH + c) * SEQ + t] = f2bf(v);
            } else if (mode == GM_OUT_F32) {
                ((float*)out)[(size_t)m * DIMX + n] = v;
            } else {
                ((u16*)out)[(size_t)m * DIMX + n] = f2bf(v);
            }
        }
    }
}

// ---------------------------------------------------------------------------
// Flash attention (causal, key/query masks, scale 1/sqrt(512)), fused residual.
// grid = (B*H, 16); block = 8 waves; each wave owns a 16-row q tile.
// ---------------------------------------------------------------------------
__global__ __launch_bounds__(256) void flash_kernel(
    const u16* __restrict__ Qb, const u16* __restrict__ Kb, const u16* __restrict__ Vt,
    const float* __restrict__ key_mask, const float* __restrict__ query_mask,
    const float* __restrict__ qn, float* __restrict__ attn_out) {
    __shared__ float sP[8][512];   // 16x32 f32 staging per wave
    const int w    = threadIdx.x >> 5;
    const int lane = threadIdx.x & 31;
    const int half = lane >> 4;
    const int ln   = lane & 15;
    const int bh   = blockIdx.x;          // b*H + h
    const int b    = bh / HEADS;
    const int h    = bh % HEADS;
    const int q0   = (blockIdx.y * 8 + w) * 16;
    const float scale = 0.04419417382415922f;   // 1/sqrt(512)

    Frag16 qf[2];   // Q A-fragments over dh chunks of 32
    {
        const u16* qp = Qb + ((size_t)bh * SEQ + q0 + ln) * DH;
#pragma unroll
        for (int c = 0; c < 2; ++c) {
            qf[c].q[0] = *(const uint4*)(qp + c * 32 + half * 8);
            qf[c].q[1] = *(const uint4*)(qp + c * 32 + 16 + half * 8);
        }
    }

    Frag8 acc[4];
#pragma unroll
    for (int i = 0; i < 4; ++i)
#pragma unroll
        for (int r = 0; r < 8; ++r) acc[i].f[r] = 0.f;
    float mrow[8], lrow[8];
#pragma unroll
    for (int r = 0; r < 8; ++r) { mrow[r] = NEGF; lrow[r] = 0.f; }

    for (int j0 = 0; j0 < q0 + 16; j0 += 32) {
        // ---- scores: S(16x32) = Q(16x64) * K_blk(32x64)^T ----
        Frag8 S[2];
#pragma unroll
        for (int sub = 0; sub < 2; ++sub) {
#pragma unroll
            for (int r = 0; r < 8; ++r) S[sub].f[r] = 0.f;
            const u16* kp = Kb + ((size_t)bh * SEQ + j0 + sub * 16 + ln) * DH;
#pragma unroll
            for (int c = 0; c < 2; ++c) {
                Frag16 kf;
                kf.q[0] = *(const uint4*)(kp + c * 32 + half * 8);
                kf.q[1] = *(const uint4*)(kp + c * 32 + 16 + half * 8);
                S[sub].v = __builtin_amdgcn_wmma_f32_16x16x32_bf16(
                    false, qf[c].v, false, kf.v, (short)0, S[sub].v, false, false);
            }
        }
        // ---- scale + key mask + causal mask; row max (16-lane shfl tree) ----
        float km0 = key_mask[b * SEQ + j0 + ln];
        float km1 = key_mask[b * SEQ + j0 + 16 + ln];
        float rmax[8];
#pragma unroll
        for (int r = 0; r < 8; ++r) {
            int q = q0 + r + half * 8;
            int k0 = j0 + ln, k1 = j0 + 16 + ln;
            float s0 = S[0].f[r] * scale, s1 = S[1].f[r] * scale;
            if (k0 > q || km0 == 0.f) s0 = NEGF;
            if (k1 > q || km1 == 0.f) s1 = NEGF;
            S[0].f[r] = s0; S[1].f[r] = s1;
            float rm = fmaxf(s0, s1);
            rm = fmaxf(rm, __shfl_xor(rm, 1, 32));
            rm = fmaxf(rm, __shfl_xor(rm, 2, 32));
            rm = fmaxf(rm, __shfl_xor(rm, 4, 32));
            rm = fmaxf(rm, __shfl_xor(rm, 8, 32));
            rmax[r] = rm;
        }
        // ---- online softmax update ----
        float alpha[8];
#pragma unroll
        for (int r = 0; r < 8; ++r) {
            float mnew = fmaxf(mrow[r], rmax[r]);
            alpha[r] = __expf(mrow[r] - mnew);
            mrow[r] = mnew;
            float p0 = __expf(S[0].f[r] - mnew);
            float p1 = __expf(S[1].f[r] - mnew);
            S[0].f[r] = p0; S[1].f[r] = p1;
            float rs = p0 + p1;
            rs += __shfl_xor(rs, 1, 32);
            rs += __shfl_xor(rs, 2, 32);
            rs += __shfl_xor(rs, 4, 32);
            rs += __shfl_xor(rs, 8, 32);
            lrow[r] = lrow[r] * alpha[r] + rs;
        }
#pragma unroll
        for (int nt = 0; nt < 4; ++nt)
#pragma unroll
            for (int r = 0; r < 8; ++r) acc[nt].f[r] *= alpha[r];

        // ---- stage P (c-frag layout) to LDS, reload as A-fragment ----
#pragma unroll
        for (int r = 0; r < 8; ++r) {
            sP[w][(r + half * 8) * 32 + ln]      = S[0].f[r];
            sP[w][(r + half * 8) * 32 + 16 + ln] = S[1].f[r];
        }
        float t0[8], t1[8];
        {
            const float* pb = &sP[w][ln * 32];
            float4 c0a = *(const float4*)(pb + half * 8);
            float4 c0b = *(const float4*)(pb + half * 8 + 4);
            float4 c1a = *(const float4*)(pb + 16 + half * 8);
            float4 c1b = *(const float4*)(pb + 16 + half * 8 + 4);
            t0[0] = c0a.x; t0[1] = c0a.y; t0[2] = c0a.z; t0[3] = c0a.w;
            t0[4] = c0b.x; t0[5] = c0b.y; t0[6] = c0b.z; t0[7] = c0b.w;
            t1[0] = c1a.x; t1[1] = c1a.y; t1[2] = c1a.z; t1[3] = c1a.w;
            t1[4] = c1b.x; t1[5] = c1b.y; t1[6] = c1b.z; t1[7] = c1b.w;
        }
        Frag16 pf;
#pragma unroll
        for (int e = 0; e < 8; ++e) { pf.u[e] = f2bf(t0[e]); pf.u[8 + e] = f2bf(t1[e]); }

        // ---- acc += P(16x32) * V_blk(32x64)  (V stored transposed [64,T]) ----
#pragma unroll
        for (int nt = 0; nt < 4; ++nt) {
            Frag16 vf;
            const u16* vp = Vt + ((size_t)bh * DH + nt * 16 + ln) * SEQ + j0 + half * 8;
            vf.q[0] = *(const uint4*)(vp);
            vf.q[1] = *(const uint4*)(vp + 16);
            acc[nt].v = __builtin_amdgcn_wmma_f32_16x16x32_bf16(
                false, pf.v, false, vf.v, (short)0, acc[nt].v, false, false);
        }
    }
    // ---- epilogue: ctx/l * query_mask + qn residual ----
#pragma unroll
    for (int r = 0; r < 8; ++r) {
        int q = q0 + r + half * 8;
        float inv = query_mask[b * SEQ + q] / lrow[r];
#pragma unroll
        for (int nt = 0; nt < 4; ++nt) {
            size_t o = ((size_t)(b * SEQ + q)) * DIMX + h * DH + nt * 16 + ln;
            attn_out[o] = acc[nt].f[r] * inv + qn[o];
        }
    }
}

// ---------------------------------------------------------------------------
// LN2 (bias-free, scale only): f32 + bf16 outputs
// ---------------------------------------------------------------------------
__global__ __launch_bounds__(256) void ln2_kernel(
    const float* __restrict__ in, const float* __restrict__ wgt,
    float* __restrict__ out_f32, u16* __restrict__ out_b) {
    int row = blockIdx.x, j = threadIdx.x;
    const float* r = in + (size_t)row * DIMX;
    float v0 = r[j], v1 = r[j + 256];
    float sum = v0 + v1, sq = v0 * v0 + v1 * v1;
    blockReduce2(sum, sq);
    float mean = sum * (1.f / DIMX);
    float var  = sq * (1.f / DIMX) - mean * mean;
    float rstd = rsqrtf(var + 1e-5f);
    float y0 = (v0 - mean) * rstd * wgt[j];
    float y1 = (v1 - mean) * rstd * wgt[j + 256];
    size_t o = (size_t)row * DIMX + j;
    out_f32[o] = y0; out_f32[o + 256] = y1;
    out_b[o] = f2bf(y0); out_b[o + 256] = f2bf(y1);
}

// ---------------------------------------------------------------------------
// final: f = LN(o + a2, ffn_w, ffn_b); out = LN(f, ln3_w)
// ---------------------------------------------------------------------------
__global__ __launch_bounds__(256) void final_ln_kernel(
    const float* __restrict__ ofc, const float* __restrict__ a2,
    const float* __restrict__ fw, const float* __restrict__ fb,
    const float* __restrict__ w3, float* __restrict__ out) {
    int row = blockIdx.x, j = threadIdx.x;
    size_t o = (size_t)row * DIMX + j;
    float y0 = ofc[o] + a2[o];
    float y1 = ofc[o + 256] + a2[o + 256];
    float sum = y0 + y1, sq = y0 * y0 + y1 * y1;
    blockReduce2(sum, sq);
    float mean = sum * (1.f / DIMX);
    float var  = sq * (1.f / DIMX) - mean * mean;
    float rstd = rsqrtf(var + 1e-5f);
    float f0 = (y0 - mean) * rstd * fw[j] + fb[j];
    float f1 = (y1 - mean) * rstd * fw[j + 256] + fb[j + 256];
    float sum2 = f0 + f1, sq2 = f0 * f0 + f1 * f1;
    blockReduce2(sum2, sq2);
    float mean2 = sum2 * (1.f / DIMX);
    float var2  = sq2 * (1.f / DIMX) - mean2 * mean2;
    float rstd2 = rsqrtf(var2 + 1e-5f);
    out[o]       = (f0 - mean2) * rstd2 * w3[j];
    out[o + 256] = (f1 - mean2) * rstd2 * w3[j + 256];
}

// ---------------------------------------------------------------------------
extern "C" void kernel_launch(void* const* d_in, const int* in_sizes, int n_in,
                              void* d_out, int out_size, void* d_ws, size_t ws_size,
                              hipStream_t stream) {
    const float* x     = (const float*)d_in[0];
    const float* ln1_w = (const float*)d_in[1];
    const float* Wq    = (const float*)d_in[2];
    const float* bq    = (const float*)d_in[3];
    const float* Wk    = (const float*)d_in[4];
    const float* bk    = (const float*)d_in[5];
    const float* Wv    = (const float*)d_in[6];
    const float* bv    = (const float*)d_in[7];
    const float* ln2_w = (const float*)d_in[8];
    const float* w1    = (const float*)d_in[9];
    const float* b1    = (const float*)d_in[10];
    const float* w2    = (const float*)d_in[11];
    const float* b2    = (const float*)d_in[12];
    const float* ffn_w = (const float*)d_in[13];
    const float* ffn_b = (const float*)d_in[14];
    const float* ln3_w = (const float*)d_in[15];

    const size_t M   = (size_t)BATCH * SEQ;              // 8192 rows
    const size_t F32 = M * DIMX * sizeof(float);         // 16 MB
    const size_t B16 = M * DIMX * sizeof(u16);           // 8 MB
    const size_t WB  = (size_t)DIMX * DIMX * sizeof(u16);

    char* p = (char*)d_ws;
    float* qn_f32  = (float*)p; p += F32;
    float* attn_o  = (float*)p; p += F32;   // attn_out, later reused for FFN output o
    float* a2_f32  = (float*)p; p += F32;
    u16*   emb_b   = (u16*)p;   p += B16;   // emb bf16, later reused for h bf16
    u16*   qn_b    = (u16*)p;   p += B16;   // qn bf16, later reused for a2 bf16
    u16*   Qb      = (u16*)p;   p += B16;
    u16*   Kb      = (u16*)p;   p += B16;
    u16*   Vt      = (u16*)p;   p += B16;
    u16*   Wq_b    = (u16*)p;   p += WB;
    u16*   Wk_b    = (u16*)p;   p += WB;
    u16*   Wv_b    = (u16*)p;   p += WB;
    u16*   w1_b    = (u16*)p;   p += WB;
    u16*   w2_b    = (u16*)p;   p += WB;
    float* key_m   = (float*)p; p += M * sizeof(float);
    float* qry_m   = (float*)p; p += M * sizeof(float);

    const int wn = DIMX * DIMX;
    cvt_bf16_kernel<<<wn / 256, 256, 0, stream>>>(Wq, Wq_b, wn);
    cvt_bf16_kernel<<<wn / 256, 256, 0, stream>>>(Wk, Wk_b, wn);
    cvt_bf16_kernel<<<wn / 256, 256, 0, stream>>>(Wv, Wv_b, wn);
    cvt_bf16_kernel<<<wn / 256, 256, 0, stream>>>(w1, w1_b, wn);
    cvt_bf16_kernel<<<wn / 256, 256, 0, stream>>>(w2, w2_b, wn);

    rope_ln1_kernel<<<(int)M, 256, 0, stream>>>(x, ln1_w, qn_f32, qn_b, emb_b,
                                                key_m, qry_m);

    dim3 gg((unsigned)(M / 128), DIMX / 64);
    gemm_nt_wmma<<<gg, 256, 0, stream>>>(qn_b,  Wq_b, bq, Qb, GM_OUT_QK, 0);
    gemm_nt_wmma<<<gg, 256, 0, stream>>>(emb_b, Wk_b, bk, Kb, GM_OUT_QK, 0);
    gemm_nt_wmma<<<gg, 256, 0, stream>>>(emb_b, Wv_b, bv, Vt, GM_OUT_VT, 0);

    flash_kernel<<<dim3(BATCH * HEADS, SEQ / 16 / 8), 256, 0, stream>>>(
        Qb, Kb, Vt, key_m, qry_m, qn_f32, attn_o);

    ln2_kernel<<<(int)M, 256, 0, stream>>>(attn_o, ln2_w, a2_f32, qn_b /*a2_b*/);

    gemm_nt_wmma<<<gg, 256, 0, stream>>>(qn_b /*a2_b*/, w1_b, b1,
                                         emb_b /*h_b*/, GM_OUT_BF16, 1);
    gemm_nt_wmma<<<gg, 256, 0, stream>>>(emb_b /*h_b*/, w2_b, b2,
                                         attn_o /*o_f32*/, GM_OUT_F32, 0);

    final_ln_kernel<<<(int)M, 256, 0, stream>>>(attn_o, a2_f32, ffn_w, ffn_b,
                                                ln3_w, (float*)d_out);
}
